// GroupedQueryAttention_33586644255008
// MI455X (gfx1250) — matrix operand order repaired and checked by reading, not compiled, not run on hardware
//
#include <hip/hip_runtime.h>
#include <hip/hip_bf16.h>

#define D_IN 2048
#define D_OUT 2048
#define N_HEADS 32
#define N_KV 8
#define HEAD_DIM 64
#define SEQ 2048

typedef __attribute__((ext_vector_type(16))) _Float16 v16h;
typedef __attribute__((ext_vector_type(8)))  _Float16 v8h;
typedef __attribute__((ext_vector_type(8)))  float    v8f;

// ---------------------------------------------------------------------------
// WMMA fragment helpers (wave32, V_WMMA_F32_16X16X32_F16)
//
// A (16x32 f16, MxK): lane L holds row M=L%16; halves h = K = g*16 + h
//                     with g = L/16.  -> two contiguous 16B chunks of a row-major row.
// B (32x16 f16, KxN): lane L holds col N=L%16; halves h = K = g*16 + h.
//                     -> one contiguous 32B chunk of an N-major (transposed) row.
// C/D (16x16 f32):    VGPR r, lane L -> row = r + g*8, col = L%16.
// ---------------------------------------------------------------------------

__device__ __forceinline__ v16h load_frag_a(const _Float16* __restrict__ base,
                                            int ld, int k0, int lane) {
  const int m = lane & 15, g = lane >> 4;
  const _Float16* p = base + (size_t)m * ld + k0 + g * 16;
  v8h lo = *(const v8h*)p;
  v8h hi = *(const v8h*)(p + 8);
  v16h r;
#pragma unroll
  for (int i = 0; i < 8; ++i) { r[i] = lo[i]; r[i + 8] = hi[i]; }
  return r;
}

__device__ __forceinline__ v16h load_frag_b(const _Float16* __restrict__ bt,
                                            int ld, int k0, int lane) {
  const int n = lane & 15, g = lane >> 4;
  return *(const v16h*)(bt + (size_t)n * ld + k0 + g * 16);
}

__device__ __forceinline__ v8f wmma_f16(v16h a, v16h b, v8f c) {
  return __builtin_amdgcn_wmma_f32_16x16x32_f16(false, a, false, b,
                                                (short)0, c, false, false);
}

// Async global->LDS copy of one 16B chunk (per-lane), tracked by ASYNCcnt.
// GV addressing: VDST = LDS byte offset, VADDR = 64-bit global address.
__device__ __forceinline__ void async_copy_b128(const _Float16* gsrc,
                                                _Float16* lds_dst) {
  unsigned lds_off = (unsigned)(size_t)lds_dst;  // low 32b of shared-aperture addr
  unsigned long long gaddr = (unsigned long long)(size_t)gsrc;
  asm volatile("global_load_async_to_lds_b128 %0, %1, off"
               :: "v"(lds_off), "v"(gaddr) : "memory");
}

// ---------------------------------------------------------------------------
// Prep kernels: f32 -> f16 convert, and f32 KxN -> f16 NxK transpose
// ---------------------------------------------------------------------------
__global__ void cvt_f32_to_f16(const float* __restrict__ src,
                               _Float16* __restrict__ dst, int n) {
  int i = blockIdx.x * 256 + threadIdx.x;
  if (i < n) dst[i] = (_Float16)src[i];
}

__global__ void transpose_f32_to_f16(const float* __restrict__ src,
                                     _Float16* __restrict__ dst, int K, int N) {
  int i = blockIdx.x * 256 + threadIdx.x;
  if (i < K * N) {
    int k = i / N, n = i - k * N;
    dst[(size_t)n * K + k] = (_Float16)src[i];
  }
}

// ---------------------------------------------------------------------------
// Fused QKV projection + RoPE with async double-buffered B-panel staging.
// grid = (16, 48), block = 256 (8 waves). Each wave: 16 rows x 64 cols; the
// whole block shares one 64-col B panel -> stage 64x32 f16 (4KB) in LDS per
// K-step via GLOBAL_LOAD_ASYNC_TO_LDS_B128 (ASYNCcnt), double buffered.
// blockIdx.y: 0..31 = Q head, 32..39 = K kv-head, 40..47 = V kv-head.
// ---------------------------------------------------------------------------
__global__ void qkv_rope_kernel(const _Float16* __restrict__ xh,
                                const _Float16* __restrict__ wq_t,
                                const _Float16* __restrict__ wk_t,
                                const _Float16* __restrict__ wv_t,
                                const float* __restrict__ cosT,
                                const float* __restrict__ sinT,
                                _Float16* __restrict__ q_h,
                                _Float16* __restrict__ k_h,
                                _Float16* __restrict__ v_t) {
  __shared__ _Float16 btile[2][64 * 32];  // [row 0..63][K 0..31], 4KB each
  const int tid  = threadIdx.x;
  const int lane = tid & 31;
  const int wv   = tid >> 5;
  const int m0   = blockIdx.x * 128 + wv * 16;
  const int sel  = blockIdx.y;
  const int g = lane >> 4, c = lane & 15;

  const _Float16* bt;
  if (sel < 32)      bt = wq_t + (size_t)sel * 64 * D_IN;
  else if (sel < 40) bt = wk_t + (size_t)(sel - 32) * 64 * D_IN;
  else               bt = wv_t + (size_t)(sel - 40) * 64 * D_IN;

  // each of the 256 threads async-copies one 16B chunk of the 4KB panel
  const int prow = tid >> 2, pch = tid & 3;

  async_copy_b128(bt + (size_t)prow * D_IN + 0 * 32 + pch * 8,
                  &btile[0][prow * 32 + pch * 8]);

  v8f acc[4] = {};
  const _Float16* arow = xh + (size_t)m0 * D_IN;

  for (int kt = 0; kt < 64; ++kt) {
    const int k0 = kt * 32;
    if (kt + 1 < 64) {
      async_copy_b128(bt + (size_t)prow * D_IN + (kt + 1) * 32 + pch * 8,
                      &btile[(kt + 1) & 1][prow * 32 + pch * 8]);
      asm volatile("s_wait_asynccnt 0x1" ::: "memory");  // tile kt landed
      __builtin_prefetch(arow + (size_t)(lane & 15) * D_IN + k0 + 64, 0, 0);
    } else {
      asm volatile("s_wait_asynccnt 0x0" ::: "memory");
    }
    __syncthreads();  // panel visible to all waves

    v16h a = load_frag_a(arow, D_IN, k0, lane);
    const _Float16* bl = &btile[kt & 1][0];
#pragma unroll
    for (int t = 0; t < 4; ++t) {
      v16h b = load_frag_b(bl + t * 16 * 32, 32, 0, lane);
      acc[t] = wmma_f16(a, b, acc[t]);
    }
    __syncthreads();  // done reading before buffer is overwritten
  }

  if (sel < 40) {  // RoPE on Q and K (head-local cols: pair d with d+32)
#pragma unroll
    for (int t = 0; t < 2; ++t)
#pragma unroll
      for (int r = 0; r < 8; ++r) {
        const int pos = m0 + r + g * 8;
        const int d = t * 16 + c;
        const float cs  = cosT[pos * HEAD_DIM + d];
        const float sn  = sinT[pos * HEAD_DIM + d];
        const float cs2 = cosT[pos * HEAD_DIM + d + 32];
        const float sn2 = sinT[pos * HEAD_DIM + d + 32];
        const float x1 = acc[t][r], x2 = acc[t + 2][r];
        acc[t][r]     = x1 * cs  - x2 * sn;
        acc[t + 2][r] = x2 * cs2 + x1 * sn2;
      }
  }

  if (sel < 32) {
    _Float16* dst = q_h + (size_t)sel * SEQ * HEAD_DIM;
#pragma unroll
    for (int t = 0; t < 4; ++t)
#pragma unroll
      for (int r = 0; r < 8; ++r) {
        const int pos = m0 + r + g * 8, d = t * 16 + c;
        dst[(size_t)pos * HEAD_DIM + d] = (_Float16)acc[t][r];
      }
  } else if (sel < 40) {
    _Float16* dst = k_h + (size_t)(sel - 32) * SEQ * HEAD_DIM;
#pragma unroll
    for (int t = 0; t < 4; ++t)
#pragma unroll
      for (int r = 0; r < 8; ++r) {
        const int pos = m0 + r + g * 8, d = t * 16 + c;
        dst[(size_t)pos * HEAD_DIM + d] = (_Float16)acc[t][r];
      }
  } else {
    _Float16* dst = v_t + (size_t)(sel - 40) * HEAD_DIM * SEQ;
#pragma unroll
    for (int t = 0; t < 4; ++t)
#pragma unroll
      for (int r = 0; r < 8; ++r) {
        const int pos = m0 + r + g * 8, d = t * 16 + c;
        dst[(size_t)d * SEQ + pos] = (_Float16)acc[t][r];
      }
  }
}

// ---------------------------------------------------------------------------
// Flash attention: one wave per (head, 16-row query tile). Keys in blocks
// of 32 so the PV update is a single K=32 WMMA per 16-col output tile.
// grid = 512 blocks x 256 threads (8 waves) = 32 heads * 128 q-tiles.
// ---------------------------------------------------------------------------
__global__ void attn_kernel(const _Float16* __restrict__ q_h,
                            const _Float16* __restrict__ k_h,
                            const _Float16* __restrict__ v_t,
                            _Float16* __restrict__ ctx_h) {
  __shared__ _Float16 plds[8][16 * 32];  // per-wave P staging (re-layout C->A)
  const int lane = threadIdx.x & 31;
  const int w    = threadIdx.x >> 5;
  const int wid  = blockIdx.x * 8 + w;
  const int head = wid >> 7;
  const int q0   = (wid & 127) * 16;
  const int kv   = head >> 2;  // 4 query heads per KV head
  const int g = lane >> 4, c = lane & 15;

  const _Float16* qbase = q_h + ((size_t)head * SEQ + q0) * HEAD_DIM;
  const v16h aq0 = load_frag_a(qbase, HEAD_DIM, 0,  lane);
  const v16h aq1 = load_frag_a(qbase, HEAD_DIM, 32, lane);

  v8f acc[4] = {};
  float m[8], l[8];
#pragma unroll
  for (int r = 0; r < 8; ++r) { m[r] = -__builtin_inff(); l[r] = 0.f; }

  for (int j0 = 0; j0 <= q0 + 15; j0 += 32) {
    // S = (Q K^T) / 8 for a 16x32 key block (two 16x16 C tiles, K-dim 64)
    v8f s[2];
#pragma unroll
    for (int st = 0; st < 2; ++st) {
      const _Float16* krow = k_h + ((size_t)kv * SEQ + j0 + st * 16) * HEAD_DIM;
      v8f z = {};
      z = wmma_f16(aq0, load_frag_b(krow, HEAD_DIM, 0,  lane), z);
      z = wmma_f16(aq1, load_frag_b(krow, HEAD_DIM, 32, lane), z);
      s[st] = z;
    }
    // scale + causal mask (elementwise in C layout)
#pragma unroll
    for (int st = 0; st < 2; ++st)
#pragma unroll
      for (int r = 0; r < 8; ++r) {
        const int i = q0 + r + g * 8, j = j0 + st * 16 + c;
        const float v = s[st][r] * 0.125f;
        s[st][r] = (j > i) ? -__builtin_inff() : v;
      }
    // online softmax: rows live in 16-lane halves -> shfl_xor reductions
#pragma unroll
    for (int r = 0; r < 8; ++r) {
      float rm = fmaxf(s[0][r], s[1][r]);
#pragma unroll
      for (int off = 1; off < 16; off <<= 1)
        rm = fmaxf(rm, __shfl_xor(rm, off, 32));
      const float mn = fmaxf(m[r], rm);
      const float sc = __expf(m[r] - mn);
      const float p0 = __expf(s[0][r] - mn);
      const float p1 = __expf(s[1][r] - mn);
      s[0][r] = p0; s[1][r] = p1;
      float rs = p0 + p1;
#pragma unroll
      for (int off = 1; off < 16; off <<= 1)
        rs += __shfl_xor(rs, off, 32);
      l[r] = l[r] * sc + rs;
      m[r] = mn;
#pragma unroll
      for (int t = 0; t < 4; ++t) acc[t][r] *= sc;
    }
    // C-layout -> A-layout re-shuffle of P through per-wave LDS
#pragma unroll
    for (int st = 0; st < 2; ++st)
#pragma unroll
      for (int r = 0; r < 8; ++r)
        plds[w][(r + g * 8) * 32 + st * 16 + c] = (_Float16)s[st][r];
    asm volatile("s_wait_dscnt 0" ::: "memory");  // wave-local LDS RAW fence
    const v16h ap = load_frag_a(&plds[w][0], 32, 0, lane);
    // O += P @ V  (V stored transposed: contiguous key dim per output col)
#pragma unroll
    for (int t = 0; t < 4; ++t) {
      v16h bv = load_frag_b(v_t + ((size_t)kv * HEAD_DIM + t * 16) * SEQ,
                            SEQ, j0, lane);
      acc[t] = wmma_f16(ap, bv, acc[t]);
    }
  }

  // normalize and scatter into ctx[n][head*64+d] (A-matrix for out-proj GEMM)
#pragma unroll
  for (int t = 0; t < 4; ++t)
#pragma unroll
    for (int r = 0; r < 8; ++r) {
      const int i = q0 + r + g * 8, d = t * 16 + c;
      ctx_h[(size_t)i * D_OUT + head * HEAD_DIM + d] =
          (_Float16)(acc[t][r] / l[r]);
    }
}

// ---------------------------------------------------------------------------
// Output projection: out = ctx @ Wo (f32 out), async-staged B panel as above.
// grid = (16, 32), block = 256.
// ---------------------------------------------------------------------------
__global__ void outproj_kernel(const _Float16* __restrict__ ctx_h,
                               const _Float16* __restrict__ wo_t,
                               float* __restrict__ out) {
  __shared__ _Float16 btile[2][64 * 32];
  const int tid  = threadIdx.x;
  const int lane = tid & 31;
  const int wv   = tid >> 5;
  const int m0   = blockIdx.x * 128 + wv * 16;
  const int n0   = blockIdx.y * 64;
  const int g = lane >> 4, c = lane & 15;

  const _Float16* bt = wo_t + (size_t)n0 * D_OUT;
  const int prow = tid >> 2, pch = tid & 3;

  async_copy_b128(bt + (size_t)prow * D_OUT + 0 * 32 + pch * 8,
                  &btile[0][prow * 32 + pch * 8]);

  v8f acc[4] = {};
  const _Float16* arow = ctx_h + (size_t)m0 * D_OUT;

  for (int kt = 0; kt < 64; ++kt) {
    const int k0 = kt * 32;
    if (kt + 1 < 64) {
      async_copy_b128(bt + (size_t)prow * D_OUT + (kt + 1) * 32 + pch * 8,
                      &btile[(kt + 1) & 1][prow * 32 + pch * 8]);
      asm volatile("s_wait_asynccnt 0x1" ::: "memory");
      __builtin_prefetch(arow + (size_t)(lane & 15) * D_OUT + k0 + 64, 0, 0);
    } else {
      asm volatile("s_wait_asynccnt 0x0" ::: "memory");
    }
    __syncthreads();

    v16h a = load_frag_a(arow, D_OUT, k0, lane);
    const _Float16* bl = &btile[kt & 1][0];
#pragma unroll
    for (int t = 0; t < 4; ++t) {
      v16h b = load_frag_b(bl + t * 16 * 32, 32, 0, lane);
      acc[t] = wmma_f16(a, b, acc[t]);
    }
    __syncthreads();
  }

#pragma unroll
  for (int t = 0; t < 4; ++t)
#pragma unroll
    for (int r = 0; r < 8; ++r) {
      const int i = m0 + r + g * 8, n = n0 + t * 16 + c;
      out[(size_t)i * D_OUT + n] = acc[t][r];
    }
}

// ---------------------------------------------------------------------------
extern "C" void kernel_launch(void* const* d_in, const int* in_sizes, int n_in,
                              void* d_out, int out_size, void* d_ws, size_t ws_size,
                              hipStream_t stream) {
  const float* x    = (const float*)d_in[0];
  const float* Wq   = (const float*)d_in[1];
  const float* Wk   = (const float*)d_in[2];
  const float* Wv   = (const float*)d_in[3];
  const float* Wo   = (const float*)d_in[4];
  const float* cosT = (const float*)d_in[5];
  const float* sinT = (const float*)d_in[6];
  float* out = (float*)d_out;

  const size_t MB = 1u << 20;
  char* ws = (char*)d_ws;
  _Float16* xh   = (_Float16*)(ws + 0 * MB);   // 2048x2048 f16  (8 MB)
  _Float16* wq_t = (_Float16*)(ws + 8 * MB);   // 2048x2048 NxK  (8 MB)
  _Float16* wk_t = (_Float16*)(ws + 16 * MB);  // 512x2048 NxK   (2 MB)
  _Float16* wv_t = (_Float16*)(ws + 18 * MB);  // 512x2048 NxK   (2 MB)
  _Float16* wo_t = (_Float16*)(ws + 20 * MB);  // 2048x2048 NxK  (8 MB)
  _Float16* q_h  = (_Float16*)(ws + 28 * MB);  // 32x2048x64     (8 MB)
  _Float16* k_h  = (_Float16*)(ws + 36 * MB);  // 8x2048x64      (2 MB)
  _Float16* v_t  = (_Float16*)(ws + 38 * MB);  // 8x64x2048      (2 MB)
  _Float16* ctx  = (_Float16*)(ws + 40 * MB);  // 2048x2048      (8 MB)

  cvt_f32_to_f16<<<16384, 256, 0, stream>>>(x, xh, D_IN * SEQ);
  transpose_f32_to_f16<<<16384, 256, 0, stream>>>(Wq, wq_t, D_IN, D_OUT);
  transpose_f32_to_f16<<<4096,  256, 0, stream>>>(Wk, wk_t, D_IN, N_KV * HEAD_DIM);
  transpose_f32_to_f16<<<4096,  256, 0, stream>>>(Wv, wv_t, D_IN, N_KV * HEAD_DIM);
  transpose_f32_to_f16<<<16384, 256, 0, stream>>>(Wo, wo_t, D_OUT, D_OUT);

  qkv_rope_kernel<<<dim3(16, 48), 256, 0, stream>>>(xh, wq_t, wk_t, wv_t,
                                                    cosT, sinT, q_h, k_h, v_t);
  attn_kernel<<<512, 256, 0, stream>>>(q_h, k_h, v_t, ctx);
  outproj_kernel<<<dim3(16, 32), 256, 0, stream>>>(ctx, wo_t, out);
}
